// CustomBertSelfAttention_8821862826506
// MI455X (gfx1250) — matrix operand order repaired and checked by reading, not compiled, not run on hardware
//
#include <hip/hip_runtime.h>
#include <hip/hip_bf16.h>

// ---------------------------------------------------------------------------
// BERT self-attention for gfx1250 (MI455X):
//   wave32 + WMMA bf16 + TDM (tensor_load_to_lds) + async-to-LDS.
//   B=2, S=2048, D=1024, H=16, HD=64
// Pipeline: f32->bf16 convert, 3x QKV GEMM (WMMA), flash-attention (WMMA).
// ---------------------------------------------------------------------------

typedef __bf16 bf16_t;
typedef __attribute__((ext_vector_type(16))) __bf16 v16bf;
typedef __attribute__((ext_vector_type(8)))  __bf16 v8bf;
typedef __attribute__((ext_vector_type(8)))  float  v8f;
typedef __attribute__((ext_vector_type(4)))  unsigned u32x4;
typedef __attribute__((ext_vector_type(8)))  unsigned u32x8;

#define NHEAD   16
#define DMODEL  1024
#define HDIM    64
#define SEQ     2048
#define BATCH   2
#define MROWS   (BATCH * SEQ)   // 4096

union Frag16 { v16bf v; uint4 u[2]; };

// ---------------------------------------------------------------------------
// CDNA5 async global->LDS copy (bypasses VGPRs, tracked by ASYNCcnt).
//   global_load_async_to_lds_b128  vdst(LDS byte addr), vaddr(64b), off
// Generic pointers to __shared__ carry the LDS byte offset in the low 32 bits
// so truncation yields the dsaddr VGPR value.
// ---------------------------------------------------------------------------
__device__ __forceinline__ void async_load_b128(const bf16_t* gsrc, bf16_t* ldst) {
  unsigned loff = (unsigned)(uintptr_t)(void*)ldst;
  unsigned long long ga = (unsigned long long)(uintptr_t)(const void*)gsrc;
  asm volatile("global_load_async_to_lds_b128 %0, %1, off"
               :: "v"(loff), "v"(ga) : "memory");
}
__device__ __forceinline__ void wait_async_all() {
  asm volatile("s_wait_asynccnt 0x0" ::: "memory");
}

// ---------------------------------------------------------------------------
// CDNA5 Tensor Data Mover: DMA a 2D tile (16-bit elements) global -> LDS.
// Builds the D# descriptor (groups 0/1, 2D form, VADDR2/3 disabled) in SGPRs
// and issues a single tensor_load_to_lds.  Tracked by TENSORcnt.
// Must be executed by exactly one wave (TDM ignores EXEC within a wave).
//   rows/cols in elements; row_stride = tensor_dim0_stride in elements.
// ---------------------------------------------------------------------------
__device__ __forceinline__ void tdm_load_tile_2d(const void* gsrc, void* ldst,
                                                 unsigned rows, unsigned cols,
                                                 unsigned row_stride) {
  unsigned long long ga = (unsigned long long)(uintptr_t)gsrc;
  unsigned ga_lo  = (unsigned)ga;
  unsigned ga_hi  = (unsigned)(ga >> 32);
  unsigned ldsoff = (unsigned)(uintptr_t)ldst;
  u32x4 g0;
  g0[0] = 0x1u;                                   // count=1, user descriptor
  g0[1] = ldsoff;                                 // lds_addr [63:32]
  g0[2] = ga_lo;                                  // global_addr[31:0]
  g0[3] = (ga_hi & 0x01FFFFFFu) | (2u << 30);     // global_addr[56:32] | type=2
  u32x8 g1;
  g1[0] = 0x10000u;                               // wg_mask=0, data_size=1 (2B)
  g1[1] = (cols & 0xFFFFu) << 16;                 // tensor_dim0[15:0]
  g1[2] = (cols >> 16) | ((rows & 0xFFFFu) << 16);// tensor_dim0 hi | tensor_dim1 lo
  g1[3] = (rows >> 16) | ((cols & 0xFFFFu) << 16);// tensor_dim1 hi | tile_dim0
  g1[4] = (rows & 0xFFFFu);                       // tile_dim1 | tile_dim2=0
  g1[5] = row_stride;                             // tensor_dim0_stride[31:0]
  g1[6] = 0;                                      // stride0 hi | dim1_stride lo
  g1[7] = 0;                                      // dim1_stride hi (unused, 2D)
  asm volatile("tensor_load_to_lds %0, %1" :: "s"(g0), "s"(g1) : "memory");
}

// ---------------------------------------------------------------------------
// Kernel 1: f32 -> bf16 conversion (8 elements / thread, vectorized)
// ---------------------------------------------------------------------------
__global__ __launch_bounds__(256)
void cvt_f32_to_bf16(const float* __restrict__ src, bf16_t* __restrict__ dst, int n8) {
  int i = blockIdx.x * blockDim.x + threadIdx.x;
  if (i >= n8) return;
  const float4* s = (const float4*)src + (size_t)i * 2;
  float4 a = s[0], b = s[1];
  v8bf o;
  o[0] = (bf16_t)a.x; o[1] = (bf16_t)a.y; o[2] = (bf16_t)a.z; o[3] = (bf16_t)a.w;
  o[4] = (bf16_t)b.x; o[5] = (bf16_t)b.y; o[6] = (bf16_t)b.z; o[7] = (bf16_t)b.w;
  ((v8bf*)dst)[i] = o;
}

// ---------------------------------------------------------------------------
// Kernel 2: Y = X @ W + bias, output stored bf16 in head-major [B,H,S,HD].
// Block = 128 threads (4 waves). Tile: M=64, N=64, K-step 32.
// X tile DMA'd by the TDM; W tile transposed through registers.
// ---------------------------------------------------------------------------
__global__ __launch_bounds__(128)
void qkv_gemm(const bf16_t* __restrict__ X, const bf16_t* __restrict__ W,
              const float* __restrict__ bias, bf16_t* __restrict__ Y) {
  __shared__ __align__(16) bf16_t Xs[64][32];   // [m][k]
  __shared__ __align__(16) bf16_t Wst[64][32];  // [n][k]  (transposed W tile)

  const int t    = threadIdx.x;
  const int wave = t >> 5;
  const int lane = t & 31;
  const int hi   = lane >> 4;
  const int ln   = lane & 15;
  const int row0 = blockIdx.x * 64;
  const int col0 = blockIdx.y * 64;

  v8f acc[4] = {};

  for (int kk = 0; kk < DMODEL; kk += 32) {
    __syncthreads();
    if (wave == 0) {   // X tile: 64x32 strided tile, one TDM op
      tdm_load_tile_2d(X + (size_t)row0 * DMODEL + kk, &Xs[0][0],
                       /*rows=*/64, /*cols=*/32, /*row_stride=*/DMODEL);
    }
    {   // W tile, transposed into LDS: kr = t/4, 16 contiguous cols per thread
      int kr = t >> 2, c = (t & 3) * 16;
      const uint4* g = (const uint4*)(W + (size_t)(kk + kr) * DMODEL + col0 + c);
      uint4 u0 = g[0], u1 = g[1];
      const bf16_t* e0 = (const bf16_t*)&u0;
      const bf16_t* e1 = (const bf16_t*)&u1;
#pragma unroll
      for (int i = 0; i < 8; ++i) Wst[c + i][kr]     = e0[i];
#pragma unroll
      for (int i = 0; i < 8; ++i) Wst[c + 8 + i][kr] = e1[i];
    }
    if (wave == 0) __builtin_amdgcn_s_wait_tensorcnt(0);
    __syncthreads();

    // A fragment: lane row = 16*wave + ln ; K elems kb..kb+15 (contiguous)
    Frag16 a;
    const int arow = wave * 16 + ln;
    const int kb   = hi * 16;
    a.u[0] = *(const uint4*)&Xs[arow][kb];
    a.u[1] = *(const uint4*)&Xs[arow][kb + 8];

#pragma unroll
    for (int nt = 0; nt < 4; ++nt) {
      // B fragment: N = 16*nt + ln ; K elems = 16*hi .. 16*hi+15 (contiguous)
      Frag16 bf;
      const bf16_t* p = &Wst[nt * 16 + ln][hi * 16];
      bf.u[0] = ((const uint4*)p)[0];
      bf.u[1] = ((const uint4*)p)[1];
      acc[nt] = __builtin_amdgcn_wmma_f32_16x16x32_bf16(
          false, a.v, false, bf.v, (short)0, acc[nt], false, false);
    }
  }

  // Epilogue: bias add, convert to bf16, store head-major [B,H,S,HD]
#pragma unroll
  for (int nt = 0; nt < 4; ++nt) {
    int col = col0 + nt * 16 + ln;
    float bb = bias[col];
    int h = col >> 6, d = col & (HDIM - 1);
#pragma unroll
    for (int r = 0; r < 8; ++r) {
      int m = row0 + wave * 16 + r + hi * 8;   // global row
      int b = m >> 11, s = m & (SEQ - 1);
      float val = acc[nt][r] + bb;
      Y[(((size_t)(b * NHEAD + h)) * SEQ + s) * HDIM + d] = (bf16_t)val;
    }
  }
}

// ---------------------------------------------------------------------------
// Kernel 3: flash attention.  One block per (b, h, 64-row q-tile).
// Block = 128 threads (4 waves); wave w owns query rows [16w,16w+16).
// Q tile via async-to-LDS, K tile via TDM, V transposed through registers.
// ---------------------------------------------------------------------------
__global__ __launch_bounds__(128)
void flash_attn(const bf16_t* __restrict__ Q, const bf16_t* __restrict__ K,
                const bf16_t* __restrict__ V, const float* __restrict__ mask,
                float* __restrict__ out) {
  __shared__ __align__(16) bf16_t Qs[64][64];    // [qrow][d]
  __shared__ __align__(16) bf16_t Ks[64][64];    // [key][d]
  __shared__ __align__(16) bf16_t Vst[64][64];   // [d][key] (transposed)
  __shared__ __align__(16) bf16_t Ps[64][64];    // [qrow][key]

  const int t    = threadIdx.x;
  const int wave = t >> 5;
  const int lane = t & 31;
  const int hi   = lane >> 4;
  const int ln   = lane & 15;

  int bid = blockIdx.x;
  const int qt = bid & 31;  bid >>= 5;
  const int h  = bid & 15;  bid >>= 4;
  const int b  = bid;

  const bf16_t* Qbase = Q + (((size_t)(b * NHEAD + h)) * SEQ + qt * 64) * HDIM;
  const bf16_t* Kbase = K + ((size_t)(b * NHEAD + h)) * SEQ * HDIM;
  const bf16_t* Vbase = V + ((size_t)(b * NHEAD + h)) * SEQ * HDIM;

  {   // load Q tile (64x64) via async DMA: row = t/2, 32 contiguous d / thread
    int r = t >> 1, c = (t & 1) * 32;
    const bf16_t* g = Qbase + (size_t)r * HDIM + c;
#pragma unroll
    for (int i = 0; i < 4; ++i)
      async_load_b128(g + 8 * i, &Qs[r][c + 8 * i]);
  }
  wait_async_all();
  __syncthreads();

  // Q A-fragments for both 32-wide d-slices, kept in registers for all k-tiles
  Frag16 aq[2];
  const int arow = wave * 16 + ln;
  const int kb   = hi * 16;
#pragma unroll
  for (int dslice = 0; dslice < 2; ++dslice) {
    aq[dslice].u[0] = *(const uint4*)&Qs[arow][dslice * 32 + kb];
    aq[dslice].u[1] = *(const uint4*)&Qs[arow][dslice * 32 + kb + 8];
  }

  v8f oacc[4] = {};
  float mrun[8], lrun[8];
#pragma unroll
  for (int r = 0; r < 8; ++r) { mrun[r] = -1e30f; lrun[r] = 0.0f; }

  const float scale = 0.125f;  // 1/sqrt(HD=64)

  for (int kt = 0; kt < SEQ / 64; ++kt) {
    __syncthreads();   // guard previous iteration's reads of Ks/Vst
    if (wave == 0) {   // K tile (64 keys x 64 d, contiguous): one TDM op
      tdm_load_tile_2d(Kbase + (size_t)(kt * 64) * HDIM, &Ks[0][0],
                       /*rows=*/64, /*cols=*/64, /*row_stride=*/HDIM);
    }
    {   // V tile, transposed to [d][key] through registers
      int r = t >> 1, c = (t & 1) * 32;
      const uint4* g = (const uint4*)(Vbase + (size_t)(kt * 64 + r) * HDIM + c);
      uint4 u[4]; u[0] = g[0]; u[1] = g[1]; u[2] = g[2]; u[3] = g[3];
      const bf16_t* e = (const bf16_t*)u;
#pragma unroll
      for (int i = 0; i < 32; ++i) Vst[c + i][r] = e[i];
    }
    if (kt + 1 < SEQ / 64) {   // hint next tiles into cache (global_prefetch_b8)
      __builtin_prefetch(Kbase + (size_t)(kt + 1) * 64 * HDIM, 0, 1);
      __builtin_prefetch(Vbase + (size_t)(kt + 1) * 64 * HDIM, 0, 1);
    }
    if (wave == 0) __builtin_amdgcn_s_wait_tensorcnt(0);
    __syncthreads();

    // ---- scores: S = Q . K^T  (8 WMMAs) ----
    v8f sacc[4] = {};
#pragma unroll
    for (int dslice = 0; dslice < 2; ++dslice) {
#pragma unroll
      for (int nt = 0; nt < 4; ++nt) {
        // B fragment: K-dim = d, N = key.  Ks[key][d] contiguous in d.
        Frag16 bk;
        const bf16_t* p = &Ks[nt * 16 + ln][dslice * 32 + hi * 16];
        bk.u[0] = ((const uint4*)p)[0];
        bk.u[1] = ((const uint4*)p)[1];
        sacc[nt] = __builtin_amdgcn_wmma_f32_16x16x32_bf16(
            false, aq[dslice].v, false, bk.v, (short)0, sacc[nt], false, false);
      }
    }

    // ---- scale + additive mask + tile row-max ----
    float mk[4];
#pragma unroll
    for (int nt = 0; nt < 4; ++nt)
      mk[nt] = mask[b * SEQ + kt * 64 + nt * 16 + ln];

    float rmax[8];
#pragma unroll
    for (int r = 0; r < 8; ++r) rmax[r] = -1e30f;
#pragma unroll
    for (int nt = 0; nt < 4; ++nt)
#pragma unroll
      for (int r = 0; r < 8; ++r) {
        float v = sacc[nt][r] * scale + mk[nt];
        sacc[nt][r] = v;
        rmax[r] = fmaxf(rmax[r], v);
      }
    // reduce across the 16 lanes of each half-wave (row stays in its half)
#pragma unroll
    for (int m = 1; m < 16; m <<= 1)
#pragma unroll
      for (int r = 0; r < 8; ++r)
        rmax[r] = fmaxf(rmax[r], __shfl_xor(rmax[r], m, 32));

    // ---- online softmax update ----
    float rsum[8];
#pragma unroll
    for (int r = 0; r < 8; ++r) {
      float mnew = fmaxf(mrun[r], rmax[r]);
      float resc = __expf(mrun[r] - mnew);
      mrun[r] = mnew;
      lrun[r] *= resc;
#pragma unroll
      for (int nt = 0; nt < 4; ++nt) oacc[nt][r] *= resc;
      rsum[r] = 0.0f;
    }
#pragma unroll
    for (int nt = 0; nt < 4; ++nt)
#pragma unroll
      for (int r = 0; r < 8; ++r) {
        float p = __expf(sacc[nt][r] - mrun[r]);
        rsum[r] += p;
        Ps[wave * 16 + r + hi * 8][nt * 16 + ln] = (bf16_t)p;  // D-layout -> LDS
      }
#pragma unroll
    for (int m = 1; m < 16; m <<= 1)
#pragma unroll
      for (int r = 0; r < 8; ++r)
        rsum[r] += __shfl_xor(rsum[r], m, 32);
#pragma unroll
    for (int r = 0; r < 8; ++r) lrun[r] += rsum[r];

    __syncthreads();   // Ps stores visible before re-reading as A fragments

    // ---- O += P . V  (8 WMMAs) ----
#pragma unroll
    for (int ks = 0; ks < 2; ++ks) {
      Frag16 ap;   // A fragment of P: row = arow, K-dim = key
      ap.u[0] = *(const uint4*)&Ps[arow][ks * 32 + kb];
      ap.u[1] = *(const uint4*)&Ps[arow][ks * 32 + kb + 8];
#pragma unroll
      for (int nt = 0; nt < 4; ++nt) {
        // B fragment of V: K-dim = key, N = d.  Vst[d][key] contiguous in key.
        Frag16 bv;
        const bf16_t* p = &Vst[nt * 16 + ln][ks * 32 + hi * 16];
        bv.u[0] = ((const uint4*)p)[0];
        bv.u[1] = ((const uint4*)p)[1];
        oacc[nt] = __builtin_amdgcn_wmma_f32_16x16x32_bf16(
            false, ap.v, false, bv.v, (short)0, oacc[nt], false, false);
      }
    }
  }

  // ---- finalize: O / l, store f32 [B,S,D] ----
#pragma unroll
  for (int nt = 0; nt < 4; ++nt) {
    int d = nt * 16 + ln;
#pragma unroll
    for (int r = 0; r < 8; ++r) {
      int srow = qt * 64 + wave * 16 + r + hi * 8;
      float val = oacc[nt][r] / lrun[r];
      out[((size_t)b * SEQ + srow) * DMODEL + h * HDIM + d] = val;
    }
  }
}

// ---------------------------------------------------------------------------
// Host launcher
// ---------------------------------------------------------------------------
extern "C" void kernel_launch(void* const* d_in, const int* in_sizes, int n_in,
                              void* d_out, int out_size, void* d_ws, size_t ws_size,
                              hipStream_t stream) {
  const float* hidden = (const float*)d_in[0];
  const float* mask   = (const float*)d_in[1];
  const float* Wq     = (const float*)d_in[2];
  const float* bq     = (const float*)d_in[3];
  const float* Wk     = (const float*)d_in[4];
  const float* bk     = (const float*)d_in[5];
  const float* Wv     = (const float*)d_in[6];
  const float* bv     = (const float*)d_in[7];
  float* out = (float*)d_out;

  // Workspace layout (bf16): X | Wq | Wk | Wv | Q | K | V  (~38 MiB total)
  bf16_t* Xbf  = (bf16_t*)d_ws;
  bf16_t* Wqbf = Xbf  + (size_t)MROWS * DMODEL;
  bf16_t* Wkbf = Wqbf + (size_t)DMODEL * DMODEL;
  bf16_t* Wvbf = Wkbf + (size_t)DMODEL * DMODEL;
  bf16_t* Qbf  = Wvbf + (size_t)DMODEL * DMODEL;
  bf16_t* Kbf  = Qbf  + (size_t)MROWS * DMODEL;
  bf16_t* Vbf  = Kbf  + (size_t)MROWS * DMODEL;

  // 1) f32 -> bf16 conversions
  int nX8 = MROWS * DMODEL / 8;     // 524288
  int nW8 = DMODEL * DMODEL / 8;    // 131072
  cvt_f32_to_bf16<<<nX8 / 256, 256, 0, stream>>>(hidden, Xbf, nX8);
  cvt_f32_to_bf16<<<nW8 / 256, 256, 0, stream>>>(Wq, Wqbf, nW8);
  cvt_f32_to_bf16<<<nW8 / 256, 256, 0, stream>>>(Wk, Wkbf, nW8);
  cvt_f32_to_bf16<<<nW8 / 256, 256, 0, stream>>>(Wv, Wvbf, nW8);

  // 2) QKV projections (WMMA GEMMs), head-major bf16 outputs
  dim3 gg(MROWS / 64, DMODEL / 64);   // 64 x 16 blocks
  qkv_gemm<<<gg, 128, 0, stream>>>(Xbf, Wqbf, bq, Qbf);
  qkv_gemm<<<gg, 128, 0, stream>>>(Xbf, Wkbf, bk, Kbf);
  qkv_gemm<<<gg, 128, 0, stream>>>(Xbf, Wvbf, bv, Vbf);

  // 3) Flash attention, f32 output [B,S,D]
  flash_attn<<<BATCH * NHEAD * (SEQ / 64), 128, 0, stream>>>(Qbf, Kbf, Vbf, mask, out);
}
